// EnhancedLightGNN_20555713479260
// MI455X (gfx1250) — matrix operand order repaired and checked by reading, not compiled, not run on hardware
//
#include <hip/hip_runtime.h>
#include <math.h>

// ---------------------------------------------------------------------------
// Types for CDNA5 WMMA (wave32)
// ---------------------------------------------------------------------------
typedef __attribute__((ext_vector_type(16))) __bf16 v16bf;
typedef __attribute__((ext_vector_type(8)))  __bf16 bf16x8;
typedef __attribute__((ext_vector_type(8)))  float  v8f;
typedef __attribute__((ext_vector_type(4)))  unsigned int u32x4;
typedef __attribute__((ext_vector_type(8)))  int i32x8;
typedef __attribute__((ext_vector_type(4)))  int i32x4;

#if defined(__HIP_DEVICE_COMPILE__) && __has_builtin(__builtin_amdgcn_tensor_load_to_lds)
#define USE_TDM 1
#else
#define USE_TDM 0
#endif

__device__ __forceinline__ float gelu_f(float x) {
    return 0.5f * x * (1.0f + erff(x * 0.70710678118654752f));
}

__device__ __forceinline__ float atomicMaxF(float* addr, float value) {
    float old = __int_as_float(__hip_atomic_load((int*)addr, __ATOMIC_RELAXED,
                                                 __HIP_MEMORY_SCOPE_AGENT));
    while (old < value) {
        int assumed = __float_as_int(old);
        int prev = atomicCAS((int*)addr, assumed, __float_as_int(value));
        if (prev == assumed) break;
        old = __int_as_float(prev);
    }
    return old;
}

// ---------------------------------------------------------------------------
// f32 -> packed bf16 with K padded to ldd (zero fill). grid.x=row, block=ldd.
// ---------------------------------------------------------------------------
__global__ void k_cvt(const float* __restrict__ src, int lds, int K,
                      __bf16* __restrict__ dst, int ldd) {
    const size_t r = blockIdx.x;
    const int c = threadIdx.x;
    float v = (c < K) ? src[r * (size_t)lds + c] : 0.0f;
    dst[r * (size_t)ldd + c] = (__bf16)v;
}

// ---------------------------------------------------------------------------
// GEMM: C[rows x O] = act( A[rows x K32]bf16 @ W[O x K32]bf16^T + bias )
// One wave per 16x16 tile, v_wmma_f32_16x16x32_bf16, K32 % 32 == 0, rows % 16 == 0.
// B tile (16 x K32) staged into LDS by the Tensor Data Mover when available.
// act: 0=none, 1=gelu, 2=tanh
// ---------------------------------------------------------------------------
__global__ __launch_bounds__(32) void k_gemm(
    const __bf16* __restrict__ A,
    const __bf16* __restrict__ Wb,
    const float* __restrict__ bias,
    float* __restrict__ C,
    int O, int K32, int act)
{
    const int lane = threadIdx.x;
    const int half = lane >> 4;
    const int m16  = lane & 15;
    const int row0 = blockIdx.x << 4;
    const int col0 = blockIdx.y << 4;
    const int n    = col0 + m16;

#if USE_TDM
    __shared__ __align__(64) __bf16 tileB[16 * 256];
    {
        // D# group0: count=1 | lds_addr | global_addr(57b) | type=2
        unsigned long long ga =
            (unsigned long long)(const void*)(Wb + (size_t)col0 * K32);
        unsigned lds_addr = (unsigned)(size_t)(&tileB[0]);
        u32x4 g0;
        g0[0] = 1u;
        g0[1] = lds_addr;
        g0[2] = (unsigned)(ga & 0xffffffffu);
        g0[3] = (unsigned)((ga >> 32) & 0x01ffffffu) | (2u << 30);
        int rowsRem = O - col0; if (rowsRem > 16) rowsRem = 16;
        // D# group1: data_size=2B; tensor_dim0=K32, tensor_dim1=rowsRem (OOB->0);
        // tile_dim0=K32, tile_dim1=16; tensor_dim0_stride=K32
        i32x8 g1;
        g1[0] = 1 << 16;                       // data_size = 1 (2 bytes)
        g1[1] = (K32 & 0xffff) << 16;          // tensor_dim0[15:0]
        g1[2] = (rowsRem & 0xffff) << 16;      // tensor_dim0[31:16]=0 | tensor_dim1[15:0]
        g1[3] = (K32 & 0xffff) << 16;          // tensor_dim1[31:16]=0 | tile_dim0
        g1[4] = 16;                            // tile_dim1 = 16, tile_dim2 = 0
        g1[5] = K32;                           // tensor_dim0_stride[31:0]
        g1[6] = 0;
        g1[7] = 0;
        i32x4 z4 = {0, 0, 0, 0};
#if __clang_major__ >= 23
        i32x8 z8 = {0, 0, 0, 0, 0, 0, 0, 0};
        __builtin_amdgcn_tensor_load_to_lds(g0, g1, z4, z4, z8, 0);
#else
        __builtin_amdgcn_tensor_load_to_lds(g0, g1, z4, z4, 0);
#endif
        __builtin_amdgcn_s_wait_tensorcnt(0);
    }
    const __bf16* Brow = tileB + (size_t)m16 * K32;
#else
    const __bf16* Brow = Wb + (size_t)((n < O) ? n : 0) * K32;
#endif

    v8f acc = {};
    const __bf16* Arow = A + (size_t)(row0 + m16) * K32;

#define GEMM_STEP(KT)                                                           \
    {                                                                           \
        bf16x8 c0 = *(const bf16x8*)(Arow + (KT) + (half << 4));                \
        bf16x8 c1 = *(const bf16x8*)(Arow + (KT) + 8 + (half << 4));            \
        v16bf af = __builtin_shufflevector(c0, c1, 0, 1, 2, 3, 4, 5, 6, 7,      \
                                           8, 9, 10, 11, 12, 13, 14, 15);       \
        v16bf bfr = *(const v16bf*)(Brow + (KT) + (half << 4));                 \
        acc = __builtin_amdgcn_wmma_f32_16x16x32_bf16(false, af, false, bfr,    \
                                                      (short)0, acc, false, false); \
    }

    int kt = 0;
    if (K32 & 32) { GEMM_STEP(0); kt = 32; }
    for (; kt < K32; kt += 64) { GEMM_STEP(kt); GEMM_STEP(kt + 32); }
#undef GEMM_STEP

    // C layout: acc[rr] -> row (row0 + half*8 + rr), col (col0 + m16)
    const int col = col0 + m16;
    if (col < O) {
        const float bv = bias ? bias[col] : 0.0f;
#pragma unroll
        for (int rr = 0; rr < 8; ++rr) {
            const int row = row0 + (half << 3) + rr;
            float v = acc[rr] + bv;
            if (act == 1)      v = gelu_f(v);
            else if (act == 2) v = tanhf(v);
            C[(size_t)row * O + col] = v;
        }
    }
}

// ---------------------------------------------------------------------------
// Row LayerNorm (wave per row): Y = ln(X (+R)) * g + b, optional gelu.
// ---------------------------------------------------------------------------
__global__ __launch_bounds__(256) void k_ln(
    const float* __restrict__ X, const float* __restrict__ R,
    const float* __restrict__ g, const float* __restrict__ b,
    float* __restrict__ Y, int rows, int D, int act)
{
    const int row  = blockIdx.x * 8 + (threadIdx.x >> 5);
    const int lane = threadIdx.x & 31;
    if (row >= rows) return;
    const float* xr = X + (size_t)row * D;
    const float* rr = R ? R + (size_t)row * D : nullptr;
    float s = 0.0f, s2 = 0.0f;
    for (int c = lane; c < D; c += 32) {
        float v = xr[c] + (rr ? rr[c] : 0.0f);
        s += v; s2 += v * v;
    }
#pragma unroll
    for (int off = 16; off; off >>= 1) {
        s  += __shfl_xor(s,  off, 32);
        s2 += __shfl_xor(s2, off, 32);
    }
    const float mu  = s / (float)D;
    const float var = s2 / (float)D - mu * mu;
    const float inv = rsqrtf(var + 1e-5f);
    float* y = Y + (size_t)row * D;
    for (int c = lane; c < D; c += 32) {
        float v = (xr[c] + (rr ? rr[c] : 0.0f) - mu) * inv * g[c] + b[c];
        if (act) v = gelu_f(v);
        y[c] = v;
    }
}

// ---------------------------------------------------------------------------
// Elementwise / scatter helpers (row = blockIdx.x, col = threadIdx.x)
// ---------------------------------------------------------------------------
__global__ void k_fill(float* p, float v, long long n) {
    long long i = blockIdx.x * (long long)blockDim.x + threadIdx.x;
    if (i < n) p[i] = v;
}
__global__ void k_axpy(float* __restrict__ y, const float* __restrict__ x,
                       float a, long long n) {
    long long i = blockIdx.x * (long long)blockDim.x + threadIdx.x;
    if (i < n) y[i] += a * x[i];
}
__global__ void k_bias_act(float* __restrict__ x, const float* __restrict__ b,
                           int act) {
    size_t i = (size_t)blockIdx.x * blockDim.x + threadIdx.x;
    float v = x[i] + (b ? b[threadIdx.x] : 0.0f);
    if (act) v = gelu_f(v);
    x[i] = v;
}
__global__ void k_count(const int* __restrict__ idx, float* __restrict__ cnt,
                        long long n) {
    long long i = blockIdx.x * (long long)blockDim.x + threadIdx.x;
    if (i < n) atomicAdd(&cnt[idx[i]], 1.0f);
}
__global__ void k_scatter_rows(const float* __restrict__ src,
                               const int* __restrict__ idx,
                               float* __restrict__ dst) {
    const size_t r = blockIdx.x;
    const int D = blockDim.x, c = threadIdx.x;
    atomicAdd(&dst[(size_t)idx[r] * D + c], src[r * D + c]);
}
__global__ void k_div_rows(float* __restrict__ x, const float* __restrict__ cnt) {
    const size_t r = blockIdx.x;
    x[r * blockDim.x + threadIdx.x] /= fmaxf(cnt[r], 1.0f);
}
__global__ void k_concat2(float* __restrict__ dst, const float* __restrict__ a,
                          const float* __restrict__ b, int Da) {
    const size_t r = blockIdx.x;
    const int D = blockDim.x, c = threadIdx.x;
    dst[r * D + c] = (c < Da) ? a[r * (size_t)Da + c]
                              : b[r * (size_t)(D - Da) + (c - Da)];
}
__global__ void k_rowdot(const float* __restrict__ X, const float* __restrict__ Wv,
                         const float* __restrict__ b, float* __restrict__ out,
                         long long rows, int D) {
    long long n = blockIdx.x * (long long)blockDim.x + threadIdx.x;
    if (n >= rows) return;
    float s = b ? b[0] : 0.0f;
    const float* xr = X + (size_t)n * D;
    for (int c = 0; c < D; ++c) s += xr[c] * Wv[c];
    out[n] = s;
}

// ---------------------------------------------------------------------------
// Segment softmax pieces: e < nReal -> d = dstIdx[e], else self-loop d=e-nReal
// hsh = log2(H)
// ---------------------------------------------------------------------------
__global__ void k_segmax(const float* __restrict__ logit, const int* __restrict__ dstIdx,
                         float* __restrict__ segMax, long long nReal, long long nLoop,
                         int hsh) {
    long long t = blockIdx.x * (long long)blockDim.x + threadIdx.x;
    const int H = 1 << hsh;
    long long tot = (nReal + nLoop) << hsh;
    if (t >= tot) return;
    long long e = t >> hsh; int hh = (int)(t & (H - 1));
    int d = (e < nReal) ? dstIdx[e] : (int)(e - nReal);
    atomicMaxF(&segMax[((size_t)d << hsh) + hh], logit[t]);
}
__global__ void k_segexp(const float* __restrict__ logit, const int* __restrict__ dstIdx,
                         const float* __restrict__ segMax, float* __restrict__ ex,
                         float* __restrict__ segSum, long long nReal, long long nLoop,
                         int hsh) {
    long long t = blockIdx.x * (long long)blockDim.x + threadIdx.x;
    const int H = 1 << hsh;
    long long tot = (nReal + nLoop) << hsh;
    if (t >= tot) return;
    long long e = t >> hsh; int hh = (int)(t & (H - 1));
    int d = (e < nReal) ? dstIdx[e] : (int)(e - nReal);
    float v = expf(logit[t] - segMax[((size_t)d << hsh) + hh]);
    ex[t] = v;
    atomicAdd(&segSum[((size_t)d << hsh) + hh], v);
}

// ---------------------------------------------------------------------------
// GATv2 logit, wave per edge. lsh = log2(lanes-per-head); H = 32 >> lsh
// ---------------------------------------------------------------------------
__global__ __launch_bounds__(32) void k_gat_logit(
    const float* __restrict__ xl, const float* __restrict__ xr,
    const float* __restrict__ efpE, const float* __restrict__ efpL,
    const int* __restrict__ src, const int* __restrict__ dst,
    const float* __restrict__ att, float* __restrict__ logit,
    long long E, int H, int Cc, int lsh) {
    const long long e = blockIdx.x;
    const int lane = threadIdx.x;
    const int lph = 32 >> (5 - lsh) ? (1 << lsh) : 1;   // lanes per head = 1<<lsh
    const int hh  = lane >> lsh;
    const int sub = lane & ((1 << lsh) - 1);
    const int cpl = Cc >> lsh;                          // channels per lane
    int s, d; const float* ef;
    if (e < E) { s = src[e]; d = dst[e]; ef = efpE + (size_t)e * H * Cc; }
    else       { s = d = (int)(e - E);  ef = efpL + (size_t)(e - E) * H * Cc; }
    const float* pl = xl + ((size_t)s * H + hh) * Cc + sub * cpl;
    const float* pr = xr + ((size_t)d * H + hh) * Cc + sub * cpl;
    const float* pe = ef + (size_t)hh * Cc + sub * cpl;
    const float* pa = att + (size_t)hh * Cc + sub * cpl;
    __builtin_prefetch(pl, 0, 1);
    __builtin_prefetch(pr, 0, 1);
    float acc = 0.0f;
    for (int c = 0; c < cpl; ++c) {
        float m = pl[c] + pr[c] + pe[c];
        m = m > 0.0f ? m : 0.2f * m;
        acc += m * pa[c];
    }
    for (int off = lph >> 1; off; off >>= 1) acc += __shfl_xor(acc, off, 32);
    if (sub == 0) logit[(e << (5 - lsh)) + hh] = acc;
}
__global__ void k_gat_scatter(const float* __restrict__ xl, const float* __restrict__ ex,
                              const float* __restrict__ segSum,
                              const int* __restrict__ src, const int* __restrict__ dst,
                              float* __restrict__ out, long long E, int H) {
    const long long e = blockIdx.x;
    const int HC = blockDim.x, c = threadIdx.x;
    const int hh = c >> 6;                  // Cc = 64
    int s, d;
    if (e < E) { s = src[e]; d = dst[e]; } else { s = d = (int)(e - E); }
    float w = ex[(size_t)e * H + hh] / segSum[(size_t)d * H + hh];
    atomicAdd(&out[(size_t)d * HC + c], w * xl[(size_t)s * HC + c]);
}

// ---------------------------------------------------------------------------
// TransformerConv (wave per edge, H=4, Cc=64: 8 lanes/head, 8 chans/lane)
// ---------------------------------------------------------------------------
__global__ __launch_bounds__(32) void k_tc_logit(
    const float* __restrict__ q, const float* __restrict__ k,
    const float* __restrict__ efp,
    const int* __restrict__ src, const int* __restrict__ dst,
    float* __restrict__ logit, float scale) {
    const long long e = blockIdx.x;
    const int lane = threadIdx.x;
    const int hh = lane >> 3, sub = lane & 7;
    const float* pq = q + ((size_t)dst[e] * 4 + hh) * 64 + sub * 8;
    const float* pk = k + ((size_t)src[e] * 4 + hh) * 64 + sub * 8;
    const float* pe = efp + ((size_t)e * 4 + hh) * 64 + sub * 8;
    __builtin_prefetch(pq, 0, 1);
    __builtin_prefetch(pk, 0, 1);
    float acc = 0.0f;
#pragma unroll
    for (int c = 0; c < 8; ++c) acc += pq[c] * (pk[c] + pe[c]);
    acc += __shfl_xor(acc, 1, 32);
    acc += __shfl_xor(acc, 2, 32);
    acc += __shfl_xor(acc, 4, 32);
    if (sub == 0) logit[e * 4 + hh] = acc * scale;
}
__global__ void k_tc_scatter(const float* __restrict__ v, const float* __restrict__ efp,
                             const float* __restrict__ ex, const float* __restrict__ segSum,
                             const int* __restrict__ src, const int* __restrict__ dst,
                             float* __restrict__ out) {
    const long long e = blockIdx.x;
    const int c = threadIdx.x;               // 256 threads
    const int hh = c >> 6;
    const int s = src[e], d = dst[e];
    float w = ex[(size_t)e * 4 + hh] / segSum[(size_t)d * 4 + hh];
    atomicAdd(&out[(size_t)d * 256 + c],
              w * (v[(size_t)s * 256 + c] + efp[(size_t)e * 256 + c]));
}
__global__ __launch_bounds__(128) void k_tc_combine(
    const float* __restrict__ acc, const float* __restrict__ xr,
    const float* __restrict__ Wb, float* __restrict__ hout, long long Nn) {
    long long n = blockIdx.x * (long long)blockDim.x + threadIdx.x;
    if (n >= Nn) return;
    float o[64];
#pragma unroll 4
    for (int c = 0; c < 64; ++c) {
        o[c] = 0.25f * (acc[(size_t)n * 256 + c]       + acc[(size_t)n * 256 + 64 + c] +
                        acc[(size_t)n * 256 + 128 + c] + acc[(size_t)n * 256 + 192 + c]);
    }
    float z = 0.0f;
    for (int c = 0; c < 64; ++c) {
        float xv = xr[(size_t)n * 64 + c];
        z += o[c] * Wb[c] + xv * Wb[64 + c] + (o[c] - xv) * Wb[128 + c];
    }
    const float beta = 1.0f / (1.0f + expf(-z));
    for (int c = 0; c < 64; ++c) {
        float xv = xr[(size_t)n * 64 + c];
        hout[(size_t)n * 64 + c] = gelu_f(beta * xv + (1.0f - beta) * o[c]);
    }
}

// ---------------------------------------------------------------------------
// Pooling scatter (block per node, 64 threads)
// ---------------------------------------------------------------------------
__global__ void k_pool_scatter(const float* __restrict__ h, const int* __restrict__ batch,
                               const float* __restrict__ ex, const float* __restrict__ gsum,
                               float* __restrict__ apool, float* __restrict__ mpool) {
    const size_t n = blockIdx.x;
    const int c = threadIdx.x;               // 64
    const int g = batch[n];
    float hv = h[n * 64 + c];
    float a = ex[n] / gsum[g];
    atomicAdd(&apool[(size_t)g * 64 + c], a * hv);
    atomicAdd(&mpool[(size_t)g * 64 + c], hv);
}

// ---------------------------------------------------------------------------
// Tiny MHA over 64 graphs, 4 heads, hd=16 (one thread per (q,head))
// ---------------------------------------------------------------------------
__global__ void k_mha(const float* __restrict__ qkv, float* __restrict__ ref,
                      int Gn, int H, int hd) {
    int t = threadIdx.x;
    if (t >= Gn * H) return;
    int qi = t / H, hh = t % H;
    const int D3 = 3 * H * hd;
    const float* q = qkv + (size_t)qi * D3 + hh * hd;
    const float scale = rsqrtf((float)hd);
    float lg[64];
    float mx = -3.0e38f;
    for (int kk = 0; kk < Gn; ++kk) {
        const float* kp = qkv + (size_t)kk * D3 + H * hd + hh * hd;
        float s = 0.0f;
        for (int d2 = 0; d2 < hd; ++d2) s += q[d2] * kp[d2];
        s *= scale;
        lg[kk] = s;
        mx = fmaxf(mx, s);
    }
    float sum = 0.0f;
    for (int kk = 0; kk < Gn; ++kk) { float e2 = expf(lg[kk] - mx); lg[kk] = e2; sum += e2; }
    const float inv = 1.0f / sum;
    for (int d2 = 0; d2 < hd; ++d2) {
        float o = 0.0f;
        for (int kk = 0; kk < Gn; ++kk)
            o += lg[kk] * qkv[(size_t)kk * D3 + 2 * H * hd + hh * hd + d2];
        ref[(size_t)qi * (H * hd) + hh * hd + d2] = o * inv;
    }
}

// ---------------------------------------------------------------------------
// Host driver
// ---------------------------------------------------------------------------
extern "C" void kernel_launch(void* const* d_in, const int* in_sizes, int n_in,
                              void* d_out, int out_size, void* d_ws, size_t ws_size,
                              hipStream_t stream) {
    (void)in_sizes; (void)n_in; (void)out_size; (void)ws_size;
    const long long N = 50000, E = 400000, G = 64;

    const float* x     = (const float*)d_in[0];
    const int*   src   = (const int*)d_in[1];
    const int*   dst   = src + E;
    const float* eattr = (const float*)d_in[2];
    const int*   batch = (const int*)d_in[3];
    const float* gfeat = (const float*)d_in[4];
    auto P = [&](int i) { return (const float*)d_in[i]; };

    float* w = (float*)d_ws;
    size_t off = 0;
    auto alloc = [&](size_t n) {                 // 32-byte aligned float alloc
        size_t nn = (n + 7) & ~(size_t)7;
        size_t o = off; off += nn; return w + o;
    };
    auto bfalloc = [&](size_t nelem) {
        return (__bf16*)alloc((nelem + 1) / 2);
    };

    float* H_   = alloc(N * 64);
    float* R_   = alloc(N * 64);
    float* T64  = alloc(N * 64);
    float* T64B = alloc(N * 64);
    float* EF   = alloc(E * 32);
    float* EAL  = alloc(N * 32);
    float* CNT  = alloc(N);
    float* P0   = alloc(N * 256);
    float* P1   = alloc(N * 256);
    float* P2   = alloc(N * 256);
    float* P3   = alloc(N * 256);
    float* EFP  = alloc(E * 256);
    float* EFPL = alloc(N * 256);
    float* LG   = alloc((E + N) * 4);
    float* EXB  = alloc((E + N) * 4);
    float* SM   = alloc(N * 4);
    float* SS   = alloc(N * 4);
    float* G32  = alloc(N * 32);
    float* GATE = alloc(N);
    float* APOOL = alloc(G * 64); float* MPOOL = alloc(G * 64); float* CNTG = alloc(G);
    float* GMAX = alloc(G); float* GSUM = alloc(G);
    float* QKV = alloc(G * 192); float* REF = alloc(G * 64); float* REFO = alloc(G * 64);
    float* POOLED = alloc(G * 128); float* GE1 = alloc(G * 128); float* GE2 = alloc(G * 64);
    float* COMB = alloc(G * 192); float* F1 = alloc(G * 256); float* F2 = alloc(G * 192);
    float* F3 = alloc(G * 128);
    // bf16 activation buffers
    __bf16* XB   = bfalloc(N * 32);
    __bf16* PEB  = bfalloc(N * 32);
    __bf16* HB   = bfalloc(N * 64);
    __bf16* EB32 = bfalloc(E * 32);
    __bf16* TE32 = bfalloc(E * 32);
    __bf16* EFB  = bfalloc(E * 32);
    __bf16* EALB = bfalloc(N * 32);
    __bf16* P3B  = bfalloc(N * 256);
    __bf16* GB   = bfalloc(G * 256);

    auto cdiv = [](long long a, long long b) { return (unsigned)((a + b - 1) / b); };
    auto cvt = [&](const float* s, int lds, int K, __bf16* dbuf, int ldd, long long rows) {
        k_cvt<<<dim3((unsigned)rows), dim3(ldd), 0, stream>>>(s, lds, K, dbuf, ldd);
    };

    // ---- convert all weights to packed bf16 (K padded to 32) --------------
    const __bf16* WB[98] = {}; int KP[98] = {};
    auto wcvt = [&](int i, int O_, int K_) {
        int Kp = (K_ + 31) & ~31;
        __bf16* pb = bfalloc((size_t)O_ * Kp);
        k_cvt<<<dim3((unsigned)O_), dim3(Kp), 0, stream>>>(P(i), K_, K_, pb, Kp);
        WB[i] = pb; KP[i] = Kp;
    };
    wcvt(5, 64, 24);   wcvt(9, 64, 64);   wcvt(13, 32, 8);  wcvt(17, 32, 32);
    wcvt(19, 64, 8);
    wcvt(20, 256, 64); wcvt(22, 256, 64); wcvt(24, 256, 32); wcvt(27, 64, 256);
    wcvt(31, 256, 64); wcvt(33, 256, 64); wcvt(35, 256, 64); wcvt(37, 256, 32);
    wcvt(39, 64, 64);
    wcvt(42, 256, 64); wcvt(44, 256, 64); wcvt(46, 256, 64); wcvt(48, 256, 32);
    wcvt(50, 64, 64);
    wcvt(53, 64, 64);  wcvt(55, 64, 64);  wcvt(57, 64, 32);
    wcvt(68, 32, 64);  wcvt(72, 192, 64); wcvt(74, 64, 64);
    wcvt(76, 128, 32); wcvt(80, 64, 128);
    wcvt(84, 256, 192); wcvt(88, 192, 256); wcvt(92, 128, 192); wcvt(96, 10, 128);

    auto gemm = [&](const __bf16* A, int wi, const float* bias, float* C,
                    long long rows, int O_, int act) {
        dim3 gr(cdiv(rows, 16), cdiv(O_, 16));
        k_gemm<<<gr, dim3(32), 0, stream>>>(A, WB[wi], bias, C, O_, KP[wi], act);
    };
    auto ln = [&](const float* X, const float* Rr, const float* g, const float* b,
                  float* Y, long long rows, int D, int act) {
        k_ln<<<cdiv(rows, 8), 256, 0, stream>>>(X, Rr, g, b, Y, (int)rows, D, act);
    };
    auto zero = [&](void* p, size_t nfl) { hipMemsetAsync(p, 0, nfl * sizeof(float), stream); };
    auto filln = [&](float* p, long long n, float v) {
        k_fill<<<cdiv(n, 256), 256, 0, stream>>>(p, v, n);
    };

    // ---- node encoder -----------------------------------------------------
    cvt(x, 24, 24, XB, 32, N);
    gemm(XB, 5, P(6), T64, N, 64, 0);
    ln(T64, nullptr, P(7), P(8), H_, N, 64, 1);
    cvt(H_, 64, 64, HB, 64, N);
    gemm(HB, 9, P(10), T64, N, 64, 0);
    ln(T64, nullptr, P(11), P(12), H_, N, 64, 1);
    cvt(x + 16, 24, 8, PEB, 32, N);
    gemm(PEB, 19, nullptr, T64, N, 64, 0);
    k_axpy<<<cdiv(N * 64, 256), 256, 0, stream>>>(H_, T64, 0.1f, N * 64);
    cvt(H_, 64, 64, HB, 64, N);

    // ---- edge encoder -----------------------------------------------------
    cvt(eattr, 8, 8, EB32, 32, E);
    gemm(EB32, 13, P(14), EFP, E, 32, 0);      // EFP reused as E x 32 f32 tmp
    ln(EFP, nullptr, P(15), P(16), EFP, E, 32, 1);
    cvt(EFP, 32, 32, TE32, 32, E);
    gemm(TE32, 17, P(18), EF, E, 32, 0);
    cvt(EF, 32, 32, EFB, 32, E);

    // ---- ea_loop ----------------------------------------------------------
    zero(CNT, N); zero(EAL, N * 32);
    k_count<<<cdiv(E, 256), 256, 0, stream>>>(dst, CNT, E);
    k_scatter_rows<<<dim3((unsigned)E), dim3(32), 0, stream>>>(EF, dst, EAL);
    k_div_rows<<<dim3((unsigned)N), dim3(32), 0, stream>>>(EAL, CNT);
    cvt(EAL, 32, 32, EALB, 32, N);

    hipMemcpyAsync(R_, H_, N * 64 * sizeof(float), hipMemcpyDeviceToDevice, stream);

    // ---- GATv2 #1 (H=4, C=64, concat) ------------------------------------
    gemm(HB, 20, P(21), P0, N, 256, 0);                 // xl
    gemm(HB, 22, P(23), P1, N, 256, 0);                 // xr
    gemm(EFB, 24, nullptr, EFP, E, 256, 0);             // ef proj
    gemm(EALB, 24, nullptr, EFPL, N, 256, 0);           // loop proj
    filln(SM, N * 4, -3.0e38f); zero(SS, N * 4); zero(P3, N * 256);
    k_gat_logit<<<dim3((unsigned)(E + N)), 32, 0, stream>>>(P0, P1, EFP, EFPL, src, dst,
                                                            P(25), LG, E, 4, 64, 3);
    k_segmax<<<cdiv((E + N) * 4, 256), 256, 0, stream>>>(LG, dst, SM, E, N, 2);
    k_segexp<<<cdiv((E + N) * 4, 256), 256, 0, stream>>>(LG, dst, SM, EXB, SS, E, N, 2);
    k_gat_scatter<<<dim3((unsigned)(E + N)), 256, 0, stream>>>(P0, EXB, SS, src, dst, P3, E, 4);
    k_bias_act<<<dim3((unsigned)N), 256, 0, stream>>>(P3, P(26), 1);
    cvt(P3, 256, 256, P3B, 256, N);
    gemm(P3B, 27, P(28), T64, N, 64, 0);
    ln(T64, nullptr, P(29), P(30), H_, N, 64, 1);
    ln(H_, R_, P(60), P(61), H_, N, 64, 0);             // n1
    hipMemcpyAsync(R_, H_, N * 64 * sizeof(float), hipMemcpyDeviceToDevice, stream);
    cvt(H_, 64, 64, HB, 64, N);

    // ---- TransformerConv layers ------------------------------------------
    for (int layer = 0; layer < 2; ++layer) {
        const int b0 = (layer == 0) ? 31 : 42;
        gemm(HB, b0 + 0, P(b0 + 1), P0, N, 256, 0);     // q
        gemm(HB, b0 + 2, P(b0 + 3), P1, N, 256, 0);     // k
        gemm(HB, b0 + 4, P(b0 + 5), P2, N, 256, 0);     // v
        gemm(EFB, b0 + 6, P(b0 + 7), EFP, E, 256, 0);   // e
        filln(SM, N * 4, -3.0e38f); zero(SS, N * 4); zero(P3, N * 256);
        k_tc_logit<<<dim3((unsigned)E), 32, 0, stream>>>(P0, P1, EFP, src, dst, LG, 0.125f);
        k_segmax<<<cdiv(E * 4, 256), 256, 0, stream>>>(LG, dst, SM, E, 0, 2);
        k_segexp<<<cdiv(E * 4, 256), 256, 0, stream>>>(LG, dst, SM, EXB, SS, E, 0, 2);
        k_tc_scatter<<<dim3((unsigned)E), 256, 0, stream>>>(P2, EFP, EXB, SS, src, dst, P3);
        gemm(HB, b0 + 8, P(b0 + 9), T64B, N, 64, 0);    // xr = h@Ws + bs
        k_tc_combine<<<cdiv(N, 128), 128, 0, stream>>>(P3, T64B, P(b0 + 10), T64, N);
        const int ng = (layer == 0) ? 62 : 64;
        ln(T64, R_, P(ng), P(ng + 1), H_, N, 64, 0);    // n2 / n3
        hipMemcpyAsync(R_, H_, N * 64 * sizeof(float), hipMemcpyDeviceToDevice, stream);
        cvt(H_, 64, 64, HB, 64, N);
    }

    // ---- GATv2 #2 (H=1, C=64, mean) --------------------------------------
    gemm(HB, 53, P(54), P0, N, 64, 0);
    gemm(HB, 55, P(56), P1, N, 64, 0);
    gemm(EFB, 57, nullptr, EFP, E, 64, 0);
    gemm(EALB, 57, nullptr, EFPL, N, 64, 0);
    filln(SM, N, -3.0e38f); zero(SS, N); zero(P3, N * 64);
    k_gat_logit<<<dim3((unsigned)(E + N)), 32, 0, stream>>>(P0, P1, EFP, EFPL, src, dst,
                                                            P(58), LG, E, 1, 64, 5);
    k_segmax<<<cdiv(E + N, 256), 256, 0, stream>>>(LG, dst, SM, E, N, 0);
    k_segexp<<<cdiv(E + N, 256), 256, 0, stream>>>(LG, dst, SM, EXB, SS, E, N, 0);
    k_gat_scatter<<<dim3((unsigned)(E + N)), 64, 0, stream>>>(P0, EXB, SS, src, dst, P3, E, 1);
    k_bias_act<<<dim3((unsigned)N), 64, 0, stream>>>(P3, P(59), 1);
    ln(P3, R_, P(66), P(67), H_, N, 64, 0);             // n4
    cvt(H_, 64, 64, HB, 64, N);

    // ---- graph pooling ----------------------------------------------------
    gemm(HB, 68, P(69), G32, N, 32, 2);                 // tanh
    k_rowdot<<<cdiv(N, 256), 256, 0, stream>>>(G32, P(70), P(71), GATE, N, 32);
    filln(GMAX, G, -3.0e38f); zero(GSUM, G);
    zero(APOOL, G * 64); zero(MPOOL, G * 64); zero(CNTG, G);
    k_segmax<<<cdiv(N, 256), 256, 0, stream>>>(GATE, batch, GMAX, N, 0, 0);
    k_segexp<<<cdiv(N, 256), 256, 0, stream>>>(GATE, batch, GMAX, EXB, GSUM, N, 0, 0);
    k_count<<<cdiv(N, 256), 256, 0, stream>>>(batch, CNTG, N);
    k_pool_scatter<<<dim3((unsigned)N), 64, 0, stream>>>(H_, batch, EXB, GSUM, APOOL, MPOOL);
    k_div_rows<<<dim3((unsigned)G), 64, 0, stream>>>(MPOOL, CNTG);

    // ---- graph MHA + heads ------------------------------------------------
    cvt(APOOL, 64, 64, GB, 64, G);
    gemm(GB, 72, P(73), QKV, G, 192, 0);
    k_mha<<<1, 256, 0, stream>>>(QKV, REF, (int)G, 4, 16);
    cvt(REF, 64, 64, GB, 64, G);
    gemm(GB, 74, P(75), REFO, G, 64, 0);
    k_concat2<<<dim3((unsigned)G), 128, 0, stream>>>(POOLED, REFO, MPOOL, 64);
    cvt(gfeat, 32, 32, GB, 32, G);
    gemm(GB, 76, P(77), GE1, G, 128, 0);
    ln(GE1, nullptr, P(78), P(79), GE1, G, 128, 1);
    cvt(GE1, 128, 128, GB, 128, G);
    gemm(GB, 80, P(81), GE2, G, 64, 0);
    ln(GE2, nullptr, P(82), P(83), GE2, G, 64, 1);
    k_concat2<<<dim3((unsigned)G), 192, 0, stream>>>(COMB, POOLED, GE2, 128);
    cvt(COMB, 192, 192, GB, 192, G);
    gemm(GB, 84, P(85), F1, G, 256, 0);
    ln(F1, nullptr, P(86), P(87), F1, G, 256, 1);
    cvt(F1, 256, 256, GB, 256, G);
    gemm(GB, 88, P(89), F2, G, 192, 0);
    ln(F2, nullptr, P(90), P(91), F2, G, 192, 1);
    cvt(F2, 192, 192, GB, 192, G);
    gemm(GB, 92, P(93), F3, G, 128, 0);
    ln(F3, nullptr, P(94), P(95), F3, G, 128, 1);

    // ---- outputs: logits (64x10) then f (64x128) --------------------------
    float* out = (float*)d_out;
    cvt(F3, 128, 128, GB, 128, G);
    gemm(GB, 96, P(97), out, G, 10, 0);
    hipMemcpyAsync(out + G * 10, F3, G * 128 * sizeof(float),
                   hipMemcpyDeviceToDevice, stream);
}
